// GNN2_2508260901137
// MI455X (gfx1250) — matrix operand order repaired and checked by reading, not compiled, not run on hardware
//
#include <hip/hip_runtime.h>
#include <hip/hip_bf16.h>
#include <stdint.h>

// ---------------------------------------------------------------------------
// Fused 3-layer GCN + MLP head for 3-node perovskite graphs, gfx1250 (CDNA5).
// One 256-thread block (8 wave32 waves) processes 16 graphs = 48 node rows.
// All activations stay in LDS; GEMMs use v_wmma_f32_16x16x32_bf16; the X tile
// is brought in with global_load_async_to_lds_b128 (ASYNCcnt path).
// HBM traffic ~= X once (38.4MB) + out (0.2MB); weights are L2-resident.
// ---------------------------------------------------------------------------

typedef __attribute__((ext_vector_type(16))) __bf16 v16bf;
typedef __attribute__((ext_vector_type(8)))  float  v8f;

#define GPB 16           // graphs per block
#define RPB 48           // node rows per block (3 * GPB)
#define HID 128
#define IND 64

union BfVec {
    uint4 q[2];
    v16bf v;
};

// A operand (16x32 bf16, M x K) from LDS row-major [row][k], stride HID.
// ISA layout: lanes 0-15 = M 0..15 with K 0..7 (v0-3) and 16..23 (v4-7);
// lanes 16-31 = same M with K 8..15 and 24..31. Per lane: two 16B loads.
static __device__ inline v16bf load_a(const __bf16* base, int rowBase, int kBase,
                                      int nl, int half) {
    BfVec t;
    const __bf16* p = base + (rowBase + nl) * HID + kBase + half * 8;
    t.q[0] = *(const uint4*)(p);
    t.q[1] = *(const uint4*)(p + 16);
    return t.v;
}

// B operand (32x16 bf16, K x N) from LDS [n][k] (transposed weights), given
// stride in k. Layout matches A: lanes 0-15 = K 0..7 and 16..23,
// lanes 16-31 = K 8..15 and 24..31, lane&15 = N. Two 16B loads.
static __device__ inline v16bf load_b(const __bf16* base, int stride, int nRow,
                                      int kBase, int half) {
    BfVec t;
    const __bf16* p = base + nRow * stride + kBase + half * 8;
    t.q[0] = *(const uint4*)(p);
    t.q[1] = *(const uint4*)(p + 16);
    return t.v;
}

// Async 16B copy global -> LDS, tracked by ASYNCcnt (GV addressing, no SADDR).
// Generic pointers to __shared__ have the LDS offset in their low 32 bits
static __device__ inline void async_copy_b128(void* lds_ptr, const void* gptr) {
    uint32_t lds_addr = (uint32_t)(uintptr_t)lds_ptr;
    uint64_t ga = (uint64_t)(uintptr_t)gptr;
    asm volatile("global_load_async_to_lds_b128 %0, %1, off"
                 :
                 : "v"(lds_addr), "v"(ga)
                 : "memory");
}

// One GCN layer: raw = Act(48xK) @ W(KxHID) via WMMA, then per-graph 3x3 mix
// + bias + relu (or mean-pool on the last layer) back into the bf16 act buf.
template <int KTOT, bool POOL>
static __device__ inline void gcn_layer(const float* __restrict__ Wg,
                                        const float* __restrict__ bg,
                                        __bf16* sA, float* sR, __bf16* sW,
                                        const float* sM,
                                        int tid, int wave, int nl, int half) {
    const v8f vzero = {0.f, 0.f, 0.f, 0.f, 0.f, 0.f, 0.f, 0.f};
    v8f acc[3];
#pragma unroll
    for (int r = 0; r < 3; ++r) acc[r] = vzero;

    const int phases = KTOT / 64;
#pragma unroll
    for (int p = 0; p < phases; ++p) {
        // Stage W k-chunk transposed: sW[n*64 + k] = W[(p*64+k)*HID + n].
        // Consecutive tid -> consecutive n -> coalesced global read.
        for (int idx = tid; idx < HID * 64; idx += 256) {
            int n = idx & 127, k = idx >> 7;
            sW[n * 64 + k] = (__bf16)Wg[(p * 64 + k) * HID + n];
        }
        __syncthreads();
#pragma unroll
        for (int kt = 0; kt < 2; ++kt) {
            v16bf bop = load_b(sW, 64, wave * 16 + nl, kt * 32, half);
#pragma unroll
            for (int r = 0; r < 3; ++r) {
                v16bf aop = load_a(sA, r * 16, p * 64 + kt * 32, nl, half);
                acc[r] = __builtin_amdgcn_wmma_f32_16x16x32_bf16(
                    false, aop, false, bop, (short)0, acc[r], false, false);
            }
        }
        __syncthreads();  // sW is restaged next phase
    }

    // Scatter f32 accumulators to LDS raw buffer.
    // C/D layout: VGPR v, lane half h -> row = 8h + v, col = lane&15.
#pragma unroll
    for (int r = 0; r < 3; ++r) {
#pragma unroll
        for (int v = 0; v < 8; ++v) {
            int row = r * 16 + half * 8 + v;
            int col = wave * 16 + nl;
            sR[row * HID + col] = acc[r][v];
        }
    }
    __syncthreads();

    // Per-graph 3x3 GCN aggregation (incl. self-loop) + bias + relu.
    for (int idx = tid; idx < GPB * HID; idx += 256) {
        int g = idx >> 7, c = idx & 127;
        float r0 = sR[(3 * g + 0) * HID + c];
        float r1 = sR[(3 * g + 1) * HID + c];
        float r2 = sR[(3 * g + 2) * HID + c];
        float bb = bg[c];
        const float* m = sM + g * 9;
        float o0 = fmaxf(m[0] * r0 + m[1] * r1 + m[2] * r2 + bb, 0.f);
        float o1 = fmaxf(m[3] * r0 + m[4] * r1 + m[5] * r2 + bb, 0.f);
        float o2 = fmaxf(m[6] * r0 + m[7] * r1 + m[8] * r2 + bb, 0.f);
        if (POOL) {
            // global mean pool over the 3 nodes -> P[g][c] in act rows 0..15
            sA[g * HID + c] = (__bf16)((o0 + o1 + o2) * (1.f / 3.f));
        } else {
            sA[(3 * g + 0) * HID + c] = (__bf16)o0;
            sA[(3 * g + 1) * HID + c] = (__bf16)o1;
            sA[(3 * g + 2) * HID + c] = (__bf16)o2;
        }
    }
    __syncthreads();
}

__global__ __launch_bounds__(256) void gnn2_fused_kernel(
    const float* __restrict__ x, const long long* __restrict__ ei,
    const float* __restrict__ W1, const float* __restrict__ b1,
    const float* __restrict__ W2, const float* __restrict__ b2,
    const float* __restrict__ W3, const float* __restrict__ b3,
    const float* __restrict__ Wm1, const float* __restrict__ bm1,
    const float* __restrict__ Wm2, const float* __restrict__ bm2,
    float* __restrict__ out, int G, long long Ntot, long long Etot) {
    __shared__ __align__(16) __bf16 sA[RPB * HID];  // activations / pooled P
    __shared__ __align__(16) float  sR[RPB * HID];  // raw GEMM f32 / X stage
    __shared__ __align__(16) __bf16 sW[128 * 64];   // staged weights [n][k]
    __shared__ float sM[GPB * 9];                   // per-graph 3x3 mix mats

    const int tid  = threadIdx.x;
    const int wave = tid >> 5;
    const int lane = tid & 31;
    const int half = lane >> 4;
    const int nl   = lane & 15;
    const int g0   = blockIdx.x * GPB;
    const long long row0 = (long long)g0 * 3;

    // Warm L2 for the deeper-layer weights (global_prefetch_b8).
    if (wave == 7) {
        __builtin_prefetch(W2 + lane * 512, 0, 0);
        __builtin_prefetch(W3 + lane * 512, 0, 0);
    }

    // ---- Kick off async copy of the X tile (contiguous 48*64 f32 span) ---
    // global_load_async_to_lds_b128 into the sR scratch (f32 staging).
    const bool fullTile = (row0 + RPB) <= Ntot;
    if (fullTile) {
        const float* src = x + row0 * IND;
        for (int i = tid; i < (RPB * IND) / 4; i += 256) {
            async_copy_b128(sR + i * 4, src + i * 4);
        }
    }

    // ---- Build per-graph 3x3 mixing matrices from edge_index -------------
    // deg[d] = 1 + #edges into d; M[d][s] = dis[d]*dis[s]*cnt(s->d)
    //                                       + (d==s) * dis[d]^2 (self loop)
    if (tid < GPB) {
        int g = g0 + tid;
        float cnt[3][3] = {};
        if (g < G) {
            long long base = (long long)12 * g;
            for (int e = 0; e < 12; ++e) {
                int s = (int)(ei[base + e] - (long long)3 * g);
                int d = (int)(ei[Etot + base + e] - (long long)3 * g);
                if (s >= 0 && s < 3 && d >= 0 && d < 3) cnt[d][s] += 1.f;
            }
        }
        float dis[3];
#pragma unroll
        for (int d = 0; d < 3; ++d)
            dis[d] = rsqrtf(1.f + cnt[d][0] + cnt[d][1] + cnt[d][2]);
#pragma unroll
        for (int d = 0; d < 3; ++d)
#pragma unroll
            for (int s = 0; s < 3; ++s)
                sM[tid * 9 + d * 3 + s] =
                    dis[d] * dis[s] * cnt[d][s] + (d == s ? dis[d] * dis[d] : 0.f);
    }

    // ---- X tile -> sA bf16 (stride HID, cols 0..63) ----------------------
    if (fullTile) {
        asm volatile("s_wait_asynccnt 0x0" ::: "memory");
        __syncthreads();  // publish async-written LDS to all waves
        const float* stage = sR;
        for (int idx = tid; idx < RPB * IND; idx += 256) {
            int r = idx >> 6, c = idx & 63;
            sA[r * HID + c] = (__bf16)stage[idx];
        }
    } else {
        // Partial tile fallback: guarded scalar loads.
        for (int idx = tid; idx < RPB * IND; idx += 256) {
            int r = idx >> 6, c = idx & 63;
            long long gr = row0 + r;
            float v = (gr < Ntot) ? x[gr * IND + c] : 0.f;
            sA[r * HID + c] = (__bf16)v;
        }
    }
    __syncthreads();

    // ---- Three GCN layers ------------------------------------------------
    gcn_layer<IND, false>(W1, b1, sA, sR, sW, sM, tid, wave, nl, half);
    gcn_layer<HID, false>(W2, b2, sA, sR, sW, sM, tid, wave, nl, half);
    gcn_layer<HID, true >(W3, b3, sA, sR, sW, sM, tid, wave, nl, half);
    // sA rows 0..15 now hold pooled P (16 x 128, bf16)

    // ---- Head layer 1: s1 = relu(P @ Wm1 + bm1), 16x128 @ 128x64 ---------
    for (int idx = tid; idx < 64 * 128; idx += 256) {
        int n = idx & 63, k = idx >> 6;
        sW[n * 128 + k] = (__bf16)Wm1[k * 64 + n];
    }
    __syncthreads();
    if (wave < 4) {  // wave-uniform branch: EXEC all-ones for WMMA
        const v8f vzero = {0.f, 0.f, 0.f, 0.f, 0.f, 0.f, 0.f, 0.f};
        v8f acc = vzero;
#pragma unroll
        for (int kt = 0; kt < 4; ++kt) {
            v16bf bop = load_b(sW, 128, wave * 16 + nl, kt * 32, half);
            v16bf aop = load_a(sA, 0, kt * 32, nl, half);
            acc = __builtin_amdgcn_wmma_f32_16x16x32_bf16(
                false, aop, false, bop, (short)0, acc, false, false);
        }
#pragma unroll
        for (int v = 0; v < 8; ++v) {
            int row = half * 8 + v;        // graph index 0..15
            int col = wave * 16 + nl;      // 0..63
            sR[row * 64 + col] = fmaxf(acc[v] + bm1[col], 0.f);
        }
    }
    __syncthreads();

    // ---- Head layer 2: out[g] = s1[g] . Wm2 + bm2 ------------------------
    if (tid < GPB) {
        int g = g0 + tid;
        if (g < G) {
            float s = bm2[0];
#pragma unroll 8
            for (int j = 0; j < 64; ++j) s += sR[tid * 64 + j] * Wm2[j];
            out[g] = s;
        }
    }
}

extern "C" void kernel_launch(void* const* d_in, const int* in_sizes, int n_in,
                              void* d_out, int out_size, void* d_ws, size_t ws_size,
                              hipStream_t stream) {
    const float*     x   = (const float*)d_in[0];
    const long long* ei  = (const long long*)d_in[1];
    // d_in[2] = batch: implied by the 3-rows-per-graph structure, unused
    const float* W1  = (const float*)d_in[3];
    const float* b1  = (const float*)d_in[4];
    const float* W2  = (const float*)d_in[5];
    const float* b2  = (const float*)d_in[6];
    const float* W3  = (const float*)d_in[7];
    const float* b3  = (const float*)d_in[8];
    const float* Wm1 = (const float*)d_in[9];
    const float* bm1 = (const float*)d_in[10];
    const float* Wm2 = (const float*)d_in[11];
    const float* bm2 = (const float*)d_in[12];
    float* out = (float*)d_out;

    const int       G    = out_size;                       // graphs
    const long long Ntot = (long long)in_sizes[0] / IND;   // nodes
    const long long Etot = (long long)in_sizes[1] / 2;     // edges

    const int blocks = (G + GPB - 1) / GPB;                // 3125 for G=50000
    gnn2_fused_kernel<<<blocks, 256, 0, stream>>>(
        x, ei, W1, b1, W2, b2, W3, b3, Wm1, bm1, Wm2, bm2, out, G, Ntot, Etot);
}
